// AttentionBlock_13864154431829
// MI455X (gfx1250) — compile-verified, matrix operand order checked by reading, not run on hardware
//
#include <hip/hip_runtime.h>
#include <hip/hip_bf16.h>
#include <math.h>

// ---------------- types ----------------
typedef __bf16 bf16_t;
typedef __attribute__((ext_vector_type(16))) __bf16 v16bf;
typedef __attribute__((ext_vector_type(8)))  __bf16 v8bf;
typedef __attribute__((ext_vector_type(8)))  float  v8f;

#define B_  8
#define L_  2048
#define E_  512
#define FF_ 2048

// Load one lane's 16-element bf16 WMMA fragment from a 32-element K-run.
// CDNA5 16-bit A/B layout: elements 0..7 -> k = hf*8 + 0..7, elements 8..15 -> k = 16 + hf*8 + 0..7
// i.e. two contiguous 16-byte chunks at +0 and +32 bytes from (run base + hf*16 bytes).
__device__ __forceinline__ v16bf load_frag16(const bf16_t* p) {
  v8bf lo = *reinterpret_cast<const v8bf*>(p);
  v8bf hi = *reinterpret_cast<const v8bf*>(p + 16);
  return __builtin_shufflevector(lo, hi, 0,1,2,3,4,5,6,7,8,9,10,11,12,13,14,15);
}

// ---------------- generic NT WMMA GEMM, async-LDS staged B ----------------
// C[b][m][n] = sum_k A[b][m][k] * W[b][n][k]  (+bias[n]) (relu?) -> f32 and/or bf16
// block = 256 threads = 8 waves; wave computes a 16x64 tile; block tile = 128x64.
// B tile (64 rows x 32 k, shared by all 8 waves) is DMA'd to LDS with
// GLOBAL_LOAD_ASYNC_TO_LDS_B128 (ASYNCcnt), double buffered; rows padded to 80B
// so ds_load_b128 fragment reads are bank-conflict free.
#define BROW 80   // 64B data + 16B pad
__global__ __launch_bounds__(256)
void gemm_bf16_nt(const bf16_t* __restrict__ A, const bf16_t* __restrict__ W,
                  const float* __restrict__ bias,
                  float* __restrict__ C32, bf16_t* __restrict__ Cbf,
                  int M, int N, int K,
                  long long sA, long long sW, long long sC, int relu)
{
  __shared__ __align__(16) char Btile[2][64 * BROW];

  const int lane = threadIdx.x & 31;
  const int wave = threadIdx.x >> 5;
  const int hf   = lane >> 4;      // K-half select (wave32 fragment striping)
  const int l16  = lane & 15;
  const int nBase = blockIdx.x * 64;
  const int mBase = blockIdx.y * 128 + wave * 16;

  const bf16_t* Ab = A + (long long)blockIdx.z * sA;
  const bf16_t* Wb = W + (long long)blockIdx.z * sW;

  const bf16_t* aPtr = Ab + (long long)(mBase + l16) * K + hf * 8;

  // cooperative async stage of one 64x32 bf16 B tile (4KB = 256 threads x 16B)
  const int srow = threadIdx.x >> 2;   // 0..63
  const int sseg = threadIdx.x & 3;    // 0..3
  const bf16_t* gStage = Wb + (long long)(nBase + srow) * K + sseg * 8;
  const unsigned ldsStage0 = (unsigned)(size_t)(&Btile[0][srow * BROW + sseg * 16]);
  const unsigned ldsStage1 = (unsigned)(size_t)(&Btile[1][srow * BROW + sseg * 16]);

  v8f acc[4];
  acc[0] = (v8f){}; acc[1] = (v8f){}; acc[2] = (v8f){}; acc[3] = (v8f){};

  const int nIter = K >> 5;

  // prologue: stage tile 0 into buffer 0
  asm volatile("global_load_async_to_lds_b128 %0, %1, off"
               :: "v"(ldsStage0), "v"(gStage) : "memory");

  for (int it = 0; it < nIter; ++it) {
    const int cur = it & 1;
    // stage next tile into the other buffer (safe: its readers finished last iter)
    if (it + 1 < nIter) {
      asm volatile("global_load_async_to_lds_b128 %0, %1, off"
                   :: "v"(cur ? ldsStage0 : ldsStage1),
                      "v"(gStage + (long long)(it + 1) * 32) : "memory");
      asm volatile("s_wait_asynccnt 0x1" ::: "memory");   // tile `it` (this wave) done
    } else {
      asm volatile("s_wait_asynccnt 0x0" ::: "memory");
    }
    __syncthreads();                                      // all waves' copies visible

    const int k0 = it * 32;
    if (it + 1 < nIter) __builtin_prefetch(aPtr + k0 + 32, 0, 1);  // A stream prefetch

    v16bf a  = load_frag16(aPtr + k0);
    const char* bb = &Btile[cur][hf * 16];
    v16bf b0 = load_frag16((const bf16_t*)(bb + (l16 +  0) * BROW));
    v16bf b1 = load_frag16((const bf16_t*)(bb + (l16 + 16) * BROW));
    v16bf b2 = load_frag16((const bf16_t*)(bb + (l16 + 32) * BROW));
    v16bf b3 = load_frag16((const bf16_t*)(bb + (l16 + 48) * BROW));
    acc[0] = __builtin_amdgcn_wmma_f32_16x16x32_bf16(false, a, false, b0, (short)0, acc[0], false, false);
    acc[1] = __builtin_amdgcn_wmma_f32_16x16x32_bf16(false, a, false, b1, (short)0, acc[1], false, false);
    acc[2] = __builtin_amdgcn_wmma_f32_16x16x32_bf16(false, a, false, b2, (short)0, acc[2], false, false);
    acc[3] = __builtin_amdgcn_wmma_f32_16x16x32_bf16(false, a, false, b3, (short)0, acc[3], false, false);

    __syncthreads();                                      // reads done before buffer reuse
  }

  // D layout: lanes 0-15 -> rows mBase+0..7, lanes 16-31 -> rows mBase+8..15; col = nBase+s*16+l16
  const long long cBase = (long long)blockIdx.z * sC;
#pragma unroll
  for (int s = 0; s < 4; ++s) {
    const int n = nBase + s * 16 + l16;
    const float bv = bias ? bias[n] : 0.0f;
#pragma unroll
    for (int r = 0; r < 8; ++r) {
      const int m = mBase + hf * 8 + r;
      float v = acc[s][r] + bv;
      if (relu) v = fmaxf(v, 0.0f);
      const long long idx = cBase + (long long)m * N + n;
      if (C32) C32[idx] = v;
      if (Cbf) Cbf[idx] = (bf16_t)v;
    }
  }
}

// ---------------- elementwise fp32 -> bf16 ----------------
__global__ __launch_bounds__(256)
void to_bf16(const float* __restrict__ s, bf16_t* __restrict__ d, long long n)
{
  long long i = (long long)blockIdx.x * 256 + threadIdx.x;
  if (i < n) d[i] = (bf16_t)s[i];
}

// ---------------- Q/K prep: slice proj, scale Q, cast bf16 ----------------
__global__ __launch_bounds__(256)
void qk_prep(const float* __restrict__ proj, bf16_t* __restrict__ q,
             bf16_t* __restrict__ k, float scale, long long n)
{
  long long i = (long long)blockIdx.x * 256 + threadIdx.x;
  if (i >= n) return;
  long long row = i >> 9;          // /512
  int col = (int)(i & 511);
  const float* pr = proj + row * (3 * E_);
  q[i] = (bf16_t)(scale * pr[col]);
  k[i] = (bf16_t)(pr[E_ + col]);
}

// ---------------- V transpose (per batch) -> Vt[b][e][s] bf16, LDS tiled ----
__global__ __launch_bounds__(256)
void v_transpose(const float* __restrict__ proj, bf16_t* __restrict__ Vt)
{
  __shared__ float tile[32][33];
  const int e0 = blockIdx.x * 32;
  const int s0 = blockIdx.y * 32;
  const int bb = blockIdx.z;
  for (int i = threadIdx.y; i < 32; i += 8) {
    const long long row = (long long)bb * L_ + (s0 + i);
    tile[i][threadIdx.x] = proj[row * (3 * E_) + 2 * E_ + e0 + threadIdx.x];
  }
  __syncthreads();
  for (int i = threadIdx.y; i < 32; i += 8) {
    const long long orow = (long long)bb * E_ + (e0 + i);
    Vt[orow * L_ + s0 + threadIdx.x] = (bf16_t)tile[threadIdx.x][i];
  }
}

// ---------------- row softmax: fp32 scores -> bf16 probs ----------------
__global__ __launch_bounds__(256)
void softmax_row_bf16(const float* __restrict__ S, bf16_t* __restrict__ P, int n)
{
  const long long row = blockIdx.x;
  const float* s = S + row * n;
  bf16_t* p = P + row * n;
  const int tid = threadIdx.x;
  __shared__ float red[8];

  float mx = -INFINITY;
  for (int i = tid; i < n; i += 256) mx = fmaxf(mx, s[i]);
  for (int off = 16; off; off >>= 1) mx = fmaxf(mx, __shfl_xor(mx, off, 32));
  if ((tid & 31) == 0) red[tid >> 5] = mx;
  __syncthreads();
  mx = red[0];
#pragma unroll
  for (int w = 1; w < 8; ++w) mx = fmaxf(mx, red[w]);
  __syncthreads();

  float sum = 0.0f;
  for (int i = tid; i < n; i += 256) sum += __expf(s[i] - mx);
  for (int off = 16; off; off >>= 1) sum += __shfl_xor(sum, off, 32);
  if ((tid & 31) == 0) red[tid >> 5] = sum;
  __syncthreads();
  sum = 0.0f;
#pragma unroll
  for (int w = 0; w < 8; ++w) sum += red[w];

  const float inv = 1.0f / sum;
  for (int i = tid; i < n; i += 256) p[i] = (bf16_t)(__expf(s[i] - mx) * inv);
}

// ---------------- residual add + LayerNorm (wave per row, E=512) ----------
__global__ __launch_bounds__(256)
void add_layernorm(const float* __restrict__ X, const float* __restrict__ Y,
                   const float* __restrict__ w, const float* __restrict__ b,
                   float* __restrict__ O32, bf16_t* __restrict__ Obf)
{
  const int lane = threadIdx.x & 31;
  const int wave = threadIdx.x >> 5;
  const long long row = (long long)blockIdx.x * 8 + wave;
  const float* px = X + row * E_;
  const float* py = Y + row * E_;

  float v[16];
  float s = 0.0f, s2 = 0.0f;
#pragma unroll
  for (int i = 0; i < 16; ++i) {
    const int c = i * 32 + lane;
    v[i] = px[c] + py[c];
    s  += v[i];
    s2 += v[i] * v[i];
  }
  for (int off = 16; off; off >>= 1) { s += __shfl_xor(s, off, 32); s2 += __shfl_xor(s2, off, 32); }
  const float mean = s * (1.0f / E_);
  const float var  = s2 * (1.0f / E_) - mean * mean;
  const float inv  = rsqrtf(var + 1e-5f);
#pragma unroll
  for (int i = 0; i < 16; ++i) {
    const int c = i * 32 + lane;
    const float o = (v[i] - mean) * inv * w[c] + b[c];
    O32[row * E_ + c] = o;
    if (Obf) Obf[row * E_ + c] = (bf16_t)o;
  }
}

// ---------------- host orchestration ----------------
extern "C" void kernel_launch(void* const* d_in, const int* in_sizes, int n_in,
                              void* d_out, int out_size, void* d_ws, size_t ws_size,
                              hipStream_t stream) {
  (void)in_sizes; (void)n_in; (void)out_size; (void)ws_size;
  const float* x    = (const float*)d_in[0];
  const float* wqkv = (const float*)d_in[1];
  const float* ln1w = (const float*)d_in[2];
  const float* ln1b = (const float*)d_in[3];
  const float* ln2w = (const float*)d_in[4];
  const float* ln2b = (const float*)d_in[5];
  const float* w1   = (const float*)d_in[6];
  const float* b1   = (const float*)d_in[7];
  const float* w2   = (const float*)d_in[8];
  const float* b2   = (const float*)d_in[9];
  float* out = (float*)d_out;

  const long long T = (long long)B_ * L_;          // 16384 tokens
  char* ws = (char*)d_ws;
  size_t off = 0;
  auto alloc = [&](size_t bytes) { size_t r = off; off += (bytes + 255) & ~(size_t)255; return r; };

  bf16_t* x_bf    = (bf16_t*)(ws + alloc(T * E_ * 2));           // 16 MB
  bf16_t* wqkv_bf = (bf16_t*)(ws + alloc((size_t)3 * E_ * E_ * 2));
  bf16_t* w1_bf   = (bf16_t*)(ws + alloc((size_t)FF_ * E_ * 2));
  bf16_t* w2_bf   = (bf16_t*)(ws + alloc((size_t)E_ * FF_ * 2));
  float*  proj    = (float*)(ws + alloc(T * 3 * E_ * 4));        // 96 MB; later aliased by P (64 MB)
  bf16_t* P       = (bf16_t*)proj;
  bf16_t* q_bf    = (bf16_t*)(ws + alloc(T * E_ * 2));
  bf16_t* k_bf    = (bf16_t*)(ws + alloc(T * E_ * 2));
  bf16_t* vt_bf   = (bf16_t*)(ws + alloc(T * E_ * 2));           // Vt: [B][E][L]
  float*  scores  = (float*)(ws + alloc((size_t)B_ * L_ * L_ * 4)); // 128 MB (fits L2); aliased by ffn1
  bf16_t* ffn1_bf = (bf16_t*)scores;
  float*  attn    = (float*)(ws + alloc(T * E_ * 4));            // 32 MB; aliased by ffn2
  float*  ffn2    = attn;
  float*  h32     = (float*)(ws + alloc(T * E_ * 4));
  bf16_t* h_bf    = (bf16_t*)(ws + alloc(T * E_ * 2));

  const float SCALE = (1.0f / sqrtf((float)E_)) * 2.0f * logf(32.0f);

  // 1) casts to bf16
  to_bf16<<<dim3((unsigned)((T * E_ + 255) / 256)), 256, 0, stream>>>(x, x_bf, T * E_);
  to_bf16<<<dim3((3 * E_ * E_ + 255) / 256), 256, 0, stream>>>(wqkv, wqkv_bf, 3LL * E_ * E_);
  to_bf16<<<dim3((FF_ * E_ + 255) / 256), 256, 0, stream>>>(w1, w1_bf, (long long)FF_ * E_);
  to_bf16<<<dim3((E_ * FF_ + 255) / 256), 256, 0, stream>>>(w2, w2_bf, (long long)E_ * FF_);

  // 2) in-projection: proj[T, 3E] = x @ Wqkv^T
  gemm_bf16_nt<<<dim3(3 * E_ / 64, (unsigned)(T / 128), 1), 256, 0, stream>>>(
      x_bf, wqkv_bf, nullptr, proj, nullptr,
      (int)T, 3 * E_, E_, 0, 0, 0, 0);

  // 3) q (scaled) / k bf16 slices; V transpose -> Vt bf16
  qk_prep<<<dim3((unsigned)((T * E_ + 255) / 256)), 256, 0, stream>>>(proj, q_bf, k_bf, SCALE, T * E_);
  v_transpose<<<dim3(E_ / 32, L_ / 32, B_), dim3(32, 8), 0, stream>>>(proj, vt_bf);

  // 4) scores[b] = (SCALE*q) @ k^T   (batched NT GEMM; 128 MB, L2-resident)
  gemm_bf16_nt<<<dim3(L_ / 64, L_ / 128, B_), 256, 0, stream>>>(
      q_bf, k_bf, nullptr, scores, nullptr,
      L_, L_, E_, (long long)L_ * E_, (long long)L_ * E_, (long long)L_ * L_, 0);

  // 5) softmax rows -> P bf16 (aliases proj)
  softmax_row_bf16<<<dim3((unsigned)T), 256, 0, stream>>>(scores, P, L_);

  // 6) attn[b] = P @ V = P @ Vt^T  (NT GEMM with contiguous K thanks to transpose)
  gemm_bf16_nt<<<dim3(E_ / 64, L_ / 128, B_), 256, 0, stream>>>(
      P, vt_bf, nullptr, attn, nullptr,
      L_, E_, L_, (long long)L_ * L_, (long long)E_ * L_, (long long)L_ * E_, 0);

  // 7) h = LN1(x + attn), keep f32 + bf16
  add_layernorm<<<dim3((unsigned)(T / 8)), 256, 0, stream>>>(x, attn, ln1w, ln1b, h32, h_bf);

  // 8) ffn hidden = relu(h @ W1^T + b1) -> bf16 only (aliases scores)
  gemm_bf16_nt<<<dim3(FF_ / 64, (unsigned)(T / 128), 1), 256, 0, stream>>>(
      h_bf, w1_bf, b1, nullptr, ffn1_bf,
      (int)T, FF_, E_, 0, 0, 0, 1);

  // 9) ffn out = hidden @ W2^T + b2 -> f32 (aliases attn)
  gemm_bf16_nt<<<dim3(E_ / 64, (unsigned)(T / 128), 1), 256, 0, stream>>>(
      ffn1_bf, w2_bf, b2, ffn2, nullptr,
      (int)T, E_, FF_, 0, 0, 0, 0);

  // 10) out = LN2(h + ffn)
  add_layernorm<<<dim3((unsigned)(T / 8)), 256, 0, stream>>>(h32, ffn2, ln2w, ln2b, out, nullptr);
}
